// PolicyNetwork_71150428225748
// MI455X (gfx1250) — matrix operand order repaired and checked by reading, not compiled, not run on hardware
//
#include <hip/hip_runtime.h>
#include <hip/hip_bf16.h>

#define N_NODES 50000
#define M_PAD   50048      // 782 * 64, padded GEMM row count
#define N_EDGES 800000
#define IN_F    32
#define EIG     64
#define HID     128
#define HEADS   4
#define DH      32

typedef __attribute__((ext_vector_type(16))) _Float16 v16h;
typedef __attribute__((ext_vector_type(8)))  _Float16 v8h;
typedef __attribute__((ext_vector_type(8)))  float    v8f;

// ---------------------------------------------------------------------------
// order-preserving uint encoding for float atomicMax
// ---------------------------------------------------------------------------
__device__ __forceinline__ unsigned ord_enc(float f) {
    unsigned b = __float_as_uint(f);
    return (b & 0x80000000u) ? ~b : (b | 0x80000000u);
}
__device__ __forceinline__ float ord_dec(unsigned u) {
    return __uint_as_float((u & 0x80000000u) ? (u & 0x7FFFFFFFu) : ~u);
}

// ---------------------------------------------------------------------------
// WMMA fragment loaders (CDNA5 wave32 16-bit layouts, CDNA5 ISA)
// A 16x32:  lanes 0-15 -> M=lane, halfs 0-7 = K 0..7, halfs 8-15 = K 16..23
//           lanes 16-31 -> same M, K base +8
// ---------------------------------------------------------------------------
__device__ __forceinline__ v16h load_a_frag(const _Float16* A, int lda,
                                            int row0, int k0, int lane) {
    int m  = lane & 15;
    int kb = (lane >> 4) << 3;          // 0 or 8
    const _Float16* p = A + (size_t)(row0 + m) * lda + k0 + kb;
    v8h lo = *(const v8h*)(p);
    v8h hi = *(const v8h*)(p + 16);
    v16h r;
#pragma unroll
    for (int e = 0; e < 8; ++e) { r[e] = lo[e]; r[e + 8] = hi[e]; }
    return r;
}

// B 32x16 from pre-transposed weights Bt[N][K] (row-major, contiguous K):
// lanes 0-15 -> N=lane, K 0..7,16..23 ; lanes 16-31 -> N=lane-16, K 8..15,24..31
__device__ __forceinline__ v16h load_b_frag(const _Float16* Bt, int ldb,
                                            int n0, int k0, int lane) {
    int n  = lane & 15;
    int kb = (lane >> 4) << 3;          // 0 or 8
    const _Float16* p = Bt + (size_t)(n0 + n) * ldb + k0 + kb;
    v8h lo = *(const v8h*)(p);
    v8h hi = *(const v8h*)(p + 16);
    v16h r;
#pragma unroll
    for (int e = 0; e < 8; ++e) { r[e] = lo[e]; r[e + 8] = hi[e]; }
    return r;
}

// ---------------------------------------------------------------------------
// Fused QKVS GEMM: C[M_PAD x 512] = A[M_PAD x K] @ Bt^T + biasCat.
// One wave per block; each wave owns a 64(M) x 64(N) tile:
// 4 A-frags x 4 B-frags -> 16 WMMAs per k-step, each fragment reused 4x.
// K is compile-time -> full unroll, loads hoistable across WMMA blocks.
// ---------------------------------------------------------------------------
template <int K>
__global__ void __launch_bounds__(32)
gemm_qkvs(const _Float16* __restrict__ A, const _Float16* __restrict__ Bt,
          const float* __restrict__ biasCat, float* __restrict__ C,
          int N) {
    int lane = threadIdx.x;
    int m_base = blockIdx.x * 64;
    int n_base = blockIdx.y * 64;

    v8f acc[4][4];
#pragma unroll
    for (int i = 0; i < 4; ++i)
#pragma unroll
        for (int j = 0; j < 4; ++j)
            acc[i][j] = (v8f){0.f,0.f,0.f,0.f,0.f,0.f,0.f,0.f};

#pragma unroll
    for (int k0 = 0; k0 < K; k0 += 32) {
        v16h a[4], b[4];
#pragma unroll
        for (int i = 0; i < 4; ++i)
            a[i] = load_a_frag(A, K, m_base + i * 16, k0, lane);
#pragma unroll
        for (int j = 0; j < 4; ++j)
            b[j] = load_b_frag(Bt, K, n_base + j * 16, k0, lane);
#pragma unroll
        for (int i = 0; i < 4; ++i)
#pragma unroll
            for (int j = 0; j < 4; ++j)
                acc[i][j] = __builtin_amdgcn_wmma_f32_16x16x32_f16(
                    false, a[i], false, b[j], (short)0, acc[i][j], false, false);
    }

    // C/D layout: lanes 0-15 -> N=lane, M=r ; lanes 16-31 -> N=lane-16, M=8+r
    int n_lane = lane & 15;
    int mhalf  = (lane >> 4) << 3;
#pragma unroll
    for (int j = 0; j < 4; ++j) {
        int col  = n_base + j * 16 + n_lane;
        float bv = biasCat[col];
#pragma unroll
        for (int i = 0; i < 4; ++i) {
            int row0 = m_base + i * 16 + mhalf;
#pragma unroll
            for (int r = 0; r < 8; ++r)
                C[(size_t)(row0 + r) * N + col] = acc[i][j][r] + bv;
        }
    }
}

// ---------------------------------------------------------------------------
// Build layer-0 A matrix (padded rows zeroed):
// xh0[i, 0:96] = f16(concat(node_feat[i], eig[sub[i]]))
// ---------------------------------------------------------------------------
__global__ void concat0_kernel(const float* __restrict__ nf,
                               const float* __restrict__ eig,
                               const int* __restrict__ sub,
                               _Float16* __restrict__ xh, int total) {
    int t = blockIdx.x * blockDim.x + threadIdx.x;
    if (t >= total) return;
    int i = t / 96, f = t - i * 96;
    float v = 0.f;
    if (i < N_NODES)
        v = (f < IN_F) ? nf[(size_t)i * IN_F + f]
                       : eig[(size_t)sub[i] * EIG + (f - IN_F)];
    xh[t] = (_Float16)v;
}

// Pack 4 weight matrices (K x 128 each) into transposed fused f16 Bt[512][K]
// and fused bias[512].
__global__ void pack_w_kernel(const float* __restrict__ wq, const float* __restrict__ wk,
                              const float* __restrict__ wv, const float* __restrict__ ws,
                              const float* __restrict__ bq, const float* __restrict__ bk,
                              const float* __restrict__ bv, const float* __restrict__ bs,
                              _Float16* __restrict__ Bt, float* __restrict__ biasCat,
                              int K) {
    int t = blockIdx.x * blockDim.x + threadIdx.x;
    if (t >= 512 * K) return;
    int n = t / K, k = t - n * K;
    const float* w = (n < 128) ? wq : (n < 256) ? wk : (n < 384) ? wv : ws;
    int nl = n & 127;
    Bt[(size_t)n * K + k] = (_Float16)w[(size_t)k * HID + nl];
    if (k == 0) {
        const float* b = (n < 128) ? bq : (n < 256) ? bk : (n < 384) ? bv : bs;
        biasCat[n] = b[nl];
    }
}

// zero segment-max (ordered-uint), denom, and attention accumulator
__global__ void init_attn_kernel(unsigned* __restrict__ smax, float* __restrict__ denom,
                                 float* __restrict__ attn, int nsm, int nattn) {
    int t = blockIdx.x * blockDim.x + threadIdx.x;
    if (t < nsm) { smax[t] = 0u; denom[t] = 0.f; }
    if (t < nattn) attn[t] = 0.f;
}

// per (edge, head): score = <q[dst,h], k[src,h]> / sqrt(DH); segment max via atomicMax
__global__ void edge_scores_kernel(const float* __restrict__ QKVS,
                                   const int* __restrict__ src, const int* __restrict__ dst,
                                   float* __restrict__ score, unsigned* __restrict__ smax,
                                   int total) {
    int t = blockIdx.x * blockDim.x + threadIdx.x;
    if (t >= total) return;
    int e = t >> 2, h = t & 3;
    int s = src[e], d = dst[e];
    const float4* q4 = (const float4*)(QKVS + (size_t)d * 512 + h * DH);
    const float4* k4 = (const float4*)(QKVS + (size_t)s * 512 + 128 + h * DH);
    float acc = 0.f;
#pragma unroll
    for (int i = 0; i < 8; ++i) {
        float4 a = q4[i], b = k4[i];
        acc += a.x * b.x + a.y * b.y + a.z * b.z + a.w * b.w;
    }
    acc *= 0.17677669529663687f;   // 1/sqrt(32)
    score[t] = acc;
    atomicMax(smax + (size_t)d * 4 + h, ord_enc(acc));
}

// e = exp(score - max[dst]); accumulate denominator
__global__ void edge_exp_kernel(const int* __restrict__ dst,
                                const unsigned* __restrict__ smax,
                                float* __restrict__ score, float* __restrict__ denom,
                                int total) {
    int t = blockIdx.x * blockDim.x + threadIdx.x;
    if (t >= total) return;
    int e = t >> 2, h = t & 3;
    int d = dst[e];
    float m  = ord_dec(smax[(size_t)d * 4 + h]);
    float ex = __expf(score[t] - m);
    score[t] = ex;
    atomicAdd(denom + (size_t)d * 4 + h, ex);
}

// alpha = e / denom[dst]
__global__ void edge_alpha_kernel(const int* __restrict__ dst,
                                  const float* __restrict__ denom,
                                  float* __restrict__ score, int total) {
    int t = blockIdx.x * blockDim.x + threadIdx.x;
    if (t >= total) return;
    int e = t >> 2, h = t & 3;
    score[t] = score[t] / denom[(size_t)dst[e] * 4 + h];
}

// attn[dst, :] += alpha[e,h] * V[src, h, :]  (one thread per edge-element)
__global__ void edge_scatter_kernel(const float* __restrict__ QKVS,
                                    const int* __restrict__ src, const int* __restrict__ dst,
                                    const float* __restrict__ alpha,
                                    float* __restrict__ attn, long total) {
    long t = (long)blockIdx.x * blockDim.x + threadIdx.x;
    if (t >= total) return;
    int e = (int)(t >> 7);
    int dco = (int)(t & 127);
    int h = dco >> 5;
    float a = alpha[(size_t)e * 4 + h];
    float v = QKVS[(size_t)src[e] * 512 + 256 + dco];
    atomicAdd(attn + (size_t)dst[e] * 128 + dco, a * v);
}

// layer-1 A matrix (padded rows zeroed):
// xh1[i,0:192] = f16(concat(attn[i]+S[i], eig[sub[i]]))
__global__ void build_x1_kernel(const float* __restrict__ attn,
                                const float* __restrict__ QKVS,
                                const float* __restrict__ eig, const int* __restrict__ sub,
                                _Float16* __restrict__ xh1, int total) {
    int t = blockIdx.x * blockDim.x + threadIdx.x;
    if (t >= total) return;
    int i = t / 192, f = t - i * 192;
    float v = 0.f;
    if (i < N_NODES)
        v = (f < 128)
            ? (attn[(size_t)i * 128 + f] + QKVS[(size_t)i * 512 + 384 + f])
            : eig[(size_t)sub[i] * EIG + (f - 128)];
    xh1[t] = (_Float16)v;
}

// MLP head on the agent node + log_softmax over num_green_phases outputs
__global__ void __launch_bounds__(128)
mlp_head_kernel(const float* __restrict__ attn, const float* __restrict__ QKVS,
                const float* __restrict__ pw1, const float* __restrict__ pb1,
                const float* __restrict__ pw2, const float* __restrict__ pb2,
                const int* __restrict__ agent_p, const int* __restrict__ ngp_p,
                float* __restrict__ out) {
    __shared__ float h[128];
    __shared__ float hid[128];
    __shared__ float z[8];
    int j = threadIdx.x;
    int a = agent_p[0];
    h[j] = attn[(size_t)a * 128 + j] + QKVS[(size_t)a * 512 + 384 + j];
    __syncthreads();
    float acc = pb1[j];
    for (int d = 0; d < 128; ++d) acc += h[d] * pw1[(size_t)d * 128 + j];
    hid[j] = (acc > 0.f) ? acc : 0.01f * acc;
    __syncthreads();
    if (j < 8) {
        float acc2 = pb2[j];
        for (int d = 0; d < 128; ++d) acc2 += hid[d] * pw2[(size_t)d * 8 + j];
        z[j] = acc2;
    }
    __syncthreads();
    if (j == 0) {
        int ngp = ngp_p[0];
        if (ngp > 8) ngp = 8;
        float m = z[0];
        for (int o = 1; o < ngp; ++o) m = fmaxf(m, z[o]);
        float se = 0.f;
        for (int o = 0; o < ngp; ++o) se += __expf(z[o] - m);
        float ls = __logf(se);
        for (int o = 0; o < ngp; ++o) out[o] = z[o] - m - ls;
    }
}

// ---------------------------------------------------------------------------
extern "C" void kernel_launch(void* const* d_in, const int* in_sizes, int n_in,
                              void* d_out, int out_size, void* d_ws, size_t ws_size,
                              hipStream_t stream) {
    const float* nf    = (const float*)d_in[0];
    const int*   ei    = (const int*)d_in[1];
    const int*   src   = ei;
    const int*   dst   = ei + N_EDGES;
    const int*   sub   = (const int*)d_in[2];
    const int*   agent = (const int*)d_in[3];
    const int*   ngp   = (const int*)d_in[4];
    const float* eig   = (const float*)d_in[5];
    const float* wq0 = (const float*)d_in[6],  *bq0 = (const float*)d_in[7];
    const float* wk0 = (const float*)d_in[8],  *bk0 = (const float*)d_in[9];
    const float* wv0 = (const float*)d_in[10], *bv0 = (const float*)d_in[11];
    const float* ws0 = (const float*)d_in[12], *bs0 = (const float*)d_in[13];
    const float* wq1 = (const float*)d_in[14], *bq1 = (const float*)d_in[15];
    const float* wk1 = (const float*)d_in[16], *bk1 = (const float*)d_in[17];
    const float* wv1 = (const float*)d_in[18], *bv1 = (const float*)d_in[19];
    const float* ws1 = (const float*)d_in[20], *bs1 = (const float*)d_in[21];
    const float* pw1 = (const float*)d_in[22], *pb1 = (const float*)d_in[23];
    const float* pw2 = (const float*)d_in[24], *pb2 = (const float*)d_in[25];
    float* out = (float*)d_out;

    // workspace carve-up (256B aligned)
    char* wsb = (char*)d_ws;
    size_t off = 0;
    auto alloc = [&](size_t bytes) -> void* {
        void* p = wsb + off;
        off = (off + bytes + 255) & ~(size_t)255;
        return p;
    };
    _Float16* xh0   = (_Float16*)alloc((size_t)M_PAD * 96 * 2);
    _Float16* xh1   = (_Float16*)alloc((size_t)M_PAD * 192 * 2);
    _Float16* W0t   = (_Float16*)alloc((size_t)512 * 96 * 2);
    _Float16* W1t   = (_Float16*)alloc((size_t)512 * 192 * 2);
    float*    bias0 = (float*)alloc(512 * 4);
    float*    bias1 = (float*)alloc(512 * 4);
    float*    QKVS  = (float*)alloc((size_t)M_PAD * 512 * 4);
    float*    attn  = (float*)alloc((size_t)N_NODES * 128 * 4);
    unsigned* smax  = (unsigned*)alloc((size_t)N_NODES * 4 * 4);
    float*    denom = (float*)alloc((size_t)N_NODES * 4 * 4);
    float*    esc   = (float*)alloc((size_t)N_EDGES * 4 * 4);
    (void)ws_size; (void)in_sizes; (void)n_in; (void)out_size;

    const int B = 256;
    int nEH   = N_EDGES * 4;                     // edge x head threads
    long nSct = (long)N_EDGES * 128;             // scatter threads
    int nsm   = N_NODES * 4;
    int nattn = N_NODES * 128;

    // ---- layer 0 ----
    {
        int tot = M_PAD * 96;
        concat0_kernel<<<(tot + B - 1) / B, B, 0, stream>>>(nf, eig, sub, xh0, tot);
        int tw = 512 * 96;
        pack_w_kernel<<<(tw + B - 1) / B, B, 0, stream>>>(wq0, wk0, wv0, ws0,
                                                          bq0, bk0, bv0, bs0,
                                                          W0t, bias0, 96);
        dim3 g(M_PAD / 64, 512 / 64);
        gemm_qkvs<96><<<g, 32, 0, stream>>>(xh0, W0t, bias0, QKVS, 512);
        init_attn_kernel<<<(nattn + B - 1) / B, B, 0, stream>>>(smax, denom, attn, nsm, nattn);
        edge_scores_kernel<<<(nEH + B - 1) / B, B, 0, stream>>>(QKVS, src, dst, esc, smax, nEH);
        edge_exp_kernel<<<(nEH + B - 1) / B, B, 0, stream>>>(dst, smax, esc, denom, nEH);
        edge_alpha_kernel<<<(nEH + B - 1) / B, B, 0, stream>>>(dst, denom, esc, nEH);
        edge_scatter_kernel<<<(unsigned)((nSct + B - 1) / B), B, 0, stream>>>(QKVS, src, dst, esc, attn, nSct);
        int t1 = M_PAD * 192;
        build_x1_kernel<<<(t1 + B - 1) / B, B, 0, stream>>>(attn, QKVS, eig, sub, xh1, t1);
    }
    // ---- layer 1 ----
    {
        int tw = 512 * 192;
        pack_w_kernel<<<(tw + B - 1) / B, B, 0, stream>>>(wq1, wk1, wv1, ws1,
                                                          bq1, bk1, bv1, bs1,
                                                          W1t, bias1, 192);
        dim3 g(M_PAD / 64, 512 / 64);
        gemm_qkvs<192><<<g, 32, 0, stream>>>(xh1, W1t, bias1, QKVS, 512);
        init_attn_kernel<<<(nattn + B - 1) / B, B, 0, stream>>>(smax, denom, attn, nsm, nattn);
        edge_scores_kernel<<<(nEH + B - 1) / B, B, 0, stream>>>(QKVS, src, dst, esc, smax, nEH);
        edge_exp_kernel<<<(nEH + B - 1) / B, B, 0, stream>>>(dst, smax, esc, denom, nEH);
        edge_alpha_kernel<<<(nEH + B - 1) / B, B, 0, stream>>>(dst, denom, esc, nEH);
        edge_scatter_kernel<<<(unsigned)((nSct + B - 1) / B), B, 0, stream>>>(QKVS, src, dst, esc, attn, nSct);
    }
    // ---- MLP head on agent node ----
    mlp_head_kernel<<<1, 128, 0, stream>>>(attn, QKVS, pw1, pb1, pw2, pb2,
                                           agent, ngp, out);
}
